// CrossAttention_89842125897795
// MI455X (gfx1250) — matrix operand order repaired and checked by reading, not compiled, not run on hardware
//
#include <hip/hip_runtime.h>
#include <hip/hip_bf16.h>

// ---------------------------------------------------------------------------
// CrossAttention on MI455X (gfx1250): bf16 WMMA for all five matmuls.
//   B=2, Tq=Tk=2048, C=1024, H=16, D=64
// ---------------------------------------------------------------------------

typedef __bf16 bf16;
typedef __attribute__((ext_vector_type(16))) __bf16 bf16x16;
typedef __attribute__((ext_vector_type(8)))  __bf16 bf16x8;
typedef __attribute__((ext_vector_type(4)))  __bf16 bf16x4;
typedef __attribute__((ext_vector_type(8)))  float  f32x8;

#define WMMA_BF16(a, b, c) \
  __builtin_amdgcn_wmma_f32_16x16x32_bf16(false, (a), false, (b), (short)0, (c), false, false)

// A-fragment (16x32 bf16, M x K) per ISA layout:
//   lane L: m = L%16, h = L/16;  elems 0..7  = A[m][8h + 0..7]
//                                elems 8..15 = A[m][16 + 8h + 0..7]
static __device__ inline bf16x16 load_a_frag(const bf16* base, int stride, int lane) {
  const int m = lane & 15, h = lane >> 4;
  union { bf16x16 v; bf16x8 p[2]; } u;
  u.p[0] = *(const bf16x8*)(base + (size_t)m * stride + 8 * h);
  u.p[1] = *(const bf16x8*)(base + (size_t)m * stride + 16 + 8 * h);
  return u.v;
}

// ---------------------------------------------------------------------------
// f32 -> bf16 conversion (vectorized x4)
// ---------------------------------------------------------------------------
__global__ void cvt_f32_to_bf16(const float* __restrict__ x, bf16* __restrict__ y, int n4) {
  int i = blockIdx.x * blockDim.x + threadIdx.x;
  if (i < n4) {
    float4 v = ((const float4*)x)[i];
    bf16x4 o;
    o[0] = (bf16)v.x; o[1] = (bf16)v.y; o[2] = (bf16)v.z; o[3] = (bf16)v.w;
    ((bf16x4*)y)[i] = o;
  }
}

// ---------------------------------------------------------------------------
// bf16 GEMM:  D[M,N] = A[M,K] * B[K,N] + bias[N]
// 256 threads = 8 waves. Block tile 128x128, BK=64 (32KB LDS).
// Waves in 2x4 grid; each wave owns 64x32 = 4x2 accumulators of 16x16.
// 16 WMMAs per barrier pair. Prefetch of the next K-tile via global_prefetch.
// ---------------------------------------------------------------------------
enum { EPI_Q = 0, EPI_KV = 1, EPI_O = 2 };

template <int EPI>
__global__ __launch_bounds__(256) void gemm_bf16_wmma(
    const bf16* __restrict__ A, const bf16* __restrict__ Bm,
    const float* __restrict__ bias,
    bf16* __restrict__ outBf, float* __restrict__ outF,
    bf16* __restrict__ outK, bf16* __restrict__ outV,
    int N, int K, int lda, int ldb, size_t aBatchStride) {
  __shared__ __attribute__((aligned(32))) bf16 sA[128 * 64];
  __shared__ __attribute__((aligned(32))) bf16 sB[64 * 128];

  const int tid  = threadIdx.x;
  const int lane = tid & 31;
  const int wave = tid >> 5;
  const int wr = wave >> 2;   // 0..1 : 64-row band
  const int wc = wave & 3;    // 0..3 : 32-col band
  const int rowBase = blockIdx.y * 128;
  const int colBase = blockIdx.x * 128;
  const int batch   = blockIdx.z;
  const bf16* Ab = A + (size_t)batch * aBatchStride;

  f32x8 acc[4][2];
#pragma unroll
  for (int i = 0; i < 4; ++i)
#pragma unroll
    for (int j = 0; j < 2; ++j)
#pragma unroll
      for (int r = 0; r < 8; ++r) acc[i][j][r] = 0.0f;

  // staging: 512 chunks of 16 bf16 (32B) each for A and B; 2 chunks/thread each
  const int aRow = tid >> 2, aCol = (tid & 3) * 16;   // rows 0..63 (+64 second)
  const int bRow = tid >> 3, bCol = (tid & 7) * 16;   // rows 0..31 (+32 second)

  for (int k0 = 0; k0 < K; k0 += 64) {
    *(bf16x16*)(sA + aRow * 64 + aCol) =
        *(const bf16x16*)(Ab + (size_t)(rowBase + aRow) * lda + k0 + aCol);
    *(bf16x16*)(sA + (64 + aRow) * 64 + aCol) =
        *(const bf16x16*)(Ab + (size_t)(rowBase + 64 + aRow) * lda + k0 + aCol);
    *(bf16x16*)(sB + bRow * 128 + bCol) =
        *(const bf16x16*)(Bm + (size_t)(k0 + bRow) * ldb + colBase + bCol);
    *(bf16x16*)(sB + (32 + bRow) * 128 + bCol) =
        *(const bf16x16*)(Bm + (size_t)(k0 + 32 + bRow) * ldb + colBase + bCol);
    if (k0 + 64 < K) {  // prefetch next K-tile (global_prefetch path)
      __builtin_prefetch(Ab + (size_t)(rowBase + aRow) * lda + k0 + 64 + aCol, 0, 1);
      __builtin_prefetch(Bm + (size_t)(k0 + 64 + bRow) * ldb + colBase + bCol, 0, 1);
    }
    __syncthreads();

    bf16x16 afrag[4][2], bfrag[2][2];
#pragma unroll
    for (int ks = 0; ks < 2; ++ks) {
#pragma unroll
      for (int i = 0; i < 4; ++i)
        afrag[i][ks] = load_a_frag(sA + (wr * 64 + i * 16) * 64 + ks * 32, 64, lane);
      // B-fragment (32x16): lane L holds column n=L%16, k in table order (same as A).
#pragma unroll
      for (int j = 0; j < 2; ++j)
#pragma unroll
        for (int s = 0; s < 16; ++s) {
          const int kk = (s < 8) ? 8 * (lane >> 4) + s : 16 + 8 * (lane >> 4) + (s - 8);
          bfrag[j][ks][s] = sB[(ks * 32 + kk) * 128 + wc * 32 + j * 16 + (lane & 15)];
        }
    }

#pragma unroll
    for (int ks = 0; ks < 2; ++ks)
#pragma unroll
      for (int i = 0; i < 4; ++i)
#pragma unroll
        for (int j = 0; j < 2; ++j)
          acc[i][j] = WMMA_BF16(afrag[i][ks], bfrag[j][ks], acc[i][j]);
    __syncthreads();
  }

  // C/D layout: VGPR r, lanes 0-15 -> (m=r, n=lane); lanes 16-31 -> (m=8+r, n=lane-16)
  const int rHalf = (lane >> 4) * 8;
  const int nLane = lane & 15;
#pragma unroll
  for (int i = 0; i < 4; ++i) {
#pragma unroll
    for (int j = 0; j < 2; ++j) {
      const int col = colBase + wc * 32 + j * 16 + nLane;
      const float bv = bias[col];
#pragma unroll
      for (int r = 0; r < 8; ++r) {
        const int row = rowBase + wr * 64 + i * 16 + rHalf + r;
        const float v = acc[i][j][r] + bv;
        if (EPI == EPI_Q) {
          outBf[(size_t)row * N + col] = (bf16)v;          // [B*Tq, C] bf16
        } else if (EPI == EPI_O) {
          outF[(size_t)row * N + col] = v;                 // [B*Tq, C] f32
        } else {  // EPI_KV: row = tk within batch; split into K^T and V
          if (col < 1024) {
            const int hh = col >> 6, dd = col & 63;
            // K^T layout [b][h][d][tk]
            outK[(((size_t)batch * 16 + hh) * 64 + dd) * 2048 + row] = (bf16)v;
          } else {
            const int c2 = col - 1024;
            const int hh = c2 >> 6, dd = c2 & 63;
            // V layout [b][h][tk][d]
            outV[((((size_t)batch * 16 + hh) * 2048 + row) * 64) + dd] = (bf16)v;
          }
        }
      }
    }
  }
}

// ---------------------------------------------------------------------------
// Flash attention: one block = 64 q-rows of one (b,h); 4 waves x 16 q-rows.
// 64-key blocks: 16 WMMAs per iteration, softmax bookkeeping (ds_bpermute
// butterflies, alpha rescale, barriers) amortized over 2x the keys.
// ---------------------------------------------------------------------------
__global__ __launch_bounds__(128) void flash_attn_wmma(
    const bf16* __restrict__ Q,   // [B][Tq][C] (head-contiguous d within row)
    const bf16* __restrict__ Kt,  // [B][H][64][Tk]
    const bf16* __restrict__ V,   // [B][H][Tk][64]
    bf16* __restrict__ O) {       // [B][Tq][C]
  __shared__ __attribute__((aligned(32))) bf16 sP[4][16 * 64];

  const int tid  = threadIdx.x;
  const int lane = tid & 31;
  const int wave = tid >> 5;
  const int bh = blockIdx.y;        // b*16 + h
  const int b = bh >> 4, h = bh & 15;
  const int qrow0 = blockIdx.x * 64 + wave * 16;

  const bf16* Qb  = Q  + ((size_t)b * 2048 + qrow0) * 1024 + h * 64;
  const bf16* Ktb = Kt + (size_t)bh * 64 * 2048;   // row d, stride Tk
  const bf16* Vb  = V  + (size_t)bh * 2048 * 64;   // row tk, stride 64

  // Q A-fragments, kept in registers for the whole key loop
  const bf16x16 aq0 = load_a_frag(Qb,      1024, lane);  // d = 0..31
  const bf16x16 aq1 = load_a_frag(Qb + 32, 1024, lane);  // d = 32..63

  float mrow[8], lrow[8];
  f32x8 o[4];
#pragma unroll
  for (int r = 0; r < 8; ++r) { mrow[r] = -1e30f; lrow[r] = 0.0f; }
#pragma unroll
  for (int d = 0; d < 4; ++d)
#pragma unroll
    for (int r = 0; r < 8; ++r) o[d][r] = 0.0f;

  f32x8 zeroc;
#pragma unroll
  for (int r = 0; r < 8; ++r) zeroc[r] = 0.0f;

  const int rHalf = (lane >> 4) * 8;
  const int nLane = lane & 15;

  for (int kb = 0; kb < 2048; kb += 64) {
    // S = Q * K^T over a 64-key block: 4 accumulators of 16 q x 16 k.
    f32x8 c[4];
#pragma unroll
    for (int g = 0; g < 4; ++g) {
      bf16x16 bk0, bk1;
#pragma unroll
      for (int s = 0; s < 16; ++s) {
        const int kk = (s < 8) ? 8 * (lane >> 4) + s : 16 + 8 * (lane >> 4) + (s - 8);
        bk0[s] = Ktb[(size_t)kk * 2048 + kb + g * 16 + nLane];
        bk1[s] = Ktb[(size_t)(32 + kk) * 2048 + kb + g * 16 + nLane];
      }
      c[g] = WMMA_BF16(aq0, bk0, zeroc);
      c[g] = WMMA_BF16(aq1, bk1, c[g]);
    }

    // online softmax update over the 64-key block
#pragma unroll
    for (int r = 0; r < 8; ++r) {
      float s0 = c[0][r] * 0.125f;   // 1/sqrt(64)
      float s1 = c[1][r] * 0.125f;
      float s2 = c[2][r] * 0.125f;
      float s3 = c[3][r] * 0.125f;
      float mv = fmaxf(fmaxf(s0, s1), fmaxf(s2, s3));
      mv = fmaxf(mv, __shfl_xor(mv, 1));
      mv = fmaxf(mv, __shfl_xor(mv, 2));
      mv = fmaxf(mv, __shfl_xor(mv, 4));
      mv = fmaxf(mv, __shfl_xor(mv, 8));
      const float mn    = fmaxf(mrow[r], mv);
      const float alpha = __expf(mrow[r] - mn);
      mrow[r] = mn;
      const float p0 = __expf(s0 - mn);
      const float p1 = __expf(s1 - mn);
      const float p2 = __expf(s2 - mn);
      const float p3 = __expf(s3 - mn);
      float ps = (p0 + p1) + (p2 + p3);
      ps += __shfl_xor(ps, 1);
      ps += __shfl_xor(ps, 2);
      ps += __shfl_xor(ps, 4);
      ps += __shfl_xor(ps, 8);
      lrow[r] = lrow[r] * alpha + ps;
#pragma unroll
      for (int d = 0; d < 4; ++d) o[d][r] *= alpha;
      // stage P (C-layout -> row-major 16x64 tile in LDS)
      bf16* row = &sP[wave][(rHalf + r) * 64 + nLane];
      row[0]  = (bf16)p0;
      row[16] = (bf16)p1;
      row[32] = (bf16)p2;
      row[48] = (bf16)p3;
    }
    __syncthreads();

    // P*V: K-contraction of 64 keys = 2 A-fragments x 4 d-blocks
    const bf16x16 ap0 = load_a_frag(&sP[wave][0],  64, lane);   // keys 0..31
    const bf16x16 ap1 = load_a_frag(&sP[wave][32], 64, lane);   // keys 32..63
#pragma unroll
    for (int d = 0; d < 4; ++d) {
      bf16x16 bv0, bv1;
#pragma unroll
      for (int s = 0; s < 16; ++s) {
        const int kk = (s < 8) ? 8 * (lane >> 4) + s : 16 + 8 * (lane >> 4) + (s - 8);
        bv0[s] = Vb[(size_t)(kb + kk) * 64 + d * 16 + nLane];
        bv1[s] = Vb[(size_t)(kb + 32 + kk) * 64 + d * 16 + nLane];
      }
      o[d] = WMMA_BF16(ap0, bv0, o[d]);
      o[d] = WMMA_BF16(ap1, bv1, o[d]);
    }
    __syncthreads();
  }

#pragma unroll
  for (int r = 0; r < 8; ++r) {
    const float inv = 1.0f / lrow[r];
    const int q = qrow0 + rHalf + r;
#pragma unroll
    for (int d = 0; d < 4; ++d)
      O[((size_t)b * 2048 + q) * 1024 + h * 64 + d * 16 + nLane] = (bf16)(o[d][r] * inv);
  }
}

// ---------------------------------------------------------------------------
// Host launcher
// ---------------------------------------------------------------------------
extern "C" void kernel_launch(void* const* d_in, const int* in_sizes, int n_in,
                              void* d_out, int out_size, void* d_ws, size_t ws_size,
                              hipStream_t stream) {
  (void)in_sizes; (void)n_in; (void)out_size; (void)ws_size;
  constexpr int B = 2, Tq = 2048, Tk = 2048, C = 1024, H = 16;

  const float* query   = (const float*)d_in[0];  // [B,Tq,C]
  const float* context = (const float*)d_in[1];  // [B,Tk,C]
  const float* Wq      = (const float*)d_in[2];  // [C,C]
  const float* bq      = (const float*)d_in[3];  // [C]
  const float* Wkv     = (const float*)d_in[4];  // [C,2C]
  const float* bkv     = (const float*)d_in[5];  // [2C]
  const float* Wo      = (const float*)d_in[6];  // [C,C]
  const float* bo      = (const float*)d_in[7];  // [C]
  float* out = (float*)d_out;                    // [B,Tq,C] f32

  // workspace carve-up (all 256B aligned)
  char* ws = (char*)d_ws;
  size_t off = 0;
  auto carve = [&](size_t bytes) {
    void* p = ws + off;
    off += (bytes + 255) & ~(size_t)255;
    return p;
  };
  bf16* qw   = (bf16*)carve((size_t)B * Tq * C * 2);       // query bf16   (reused as attn)
  bf16* cw   = (bf16*)carve((size_t)B * Tk * C * 2);       // context bf16
  bf16* wqb  = (bf16*)carve((size_t)C * C * 2);            // Wq bf16
  bf16* wkvb = (bf16*)carve((size_t)C * 2 * C * 2);        // Wkv bf16
  bf16* wob  = (bf16*)carve((size_t)C * C * 2);            // Wo bf16
  bf16* Qp   = (bf16*)carve((size_t)B * Tq * C * 2);       // Q projection
  bf16* Kt   = (bf16*)carve((size_t)B * H * 64 * Tk * 2);  // K^T [b][h][d][tk]
  bf16* Vp   = (bf16*)carve((size_t)B * H * Tk * 64 * 2);  // V   [b][h][tk][d]
  bf16* attn = qw;                                         // reuse: attn out [B,Tq,C]

  auto cvt = [&](const float* src, bf16* dst, size_t n) {
    const int n4 = (int)(n / 4);
    cvt_f32_to_bf16<<<(n4 + 255) / 256, 256, 0, stream>>>(src, dst, n4);
  };
  cvt(query,   qw,   (size_t)B * Tq * C);
  cvt(context, cw,   (size_t)B * Tk * C);
  cvt(Wq,      wqb,  (size_t)C * C);
  cvt(Wkv,     wkvb, (size_t)C * 2 * C);
  cvt(Wo,      wob,  (size_t)C * C);

  // Q = query @ Wq + bq  (flat M = B*Tq = 4096, N = 1024, K = 1024)
  gemm_bf16_wmma<EPI_Q><<<dim3(C / 128, (B * Tq) / 128, 1), 256, 0, stream>>>(
      qw, wqb, bq, Qp, nullptr, nullptr, nullptr, C, C, C, C, 0);

  // KV = context @ Wkv + bkv  (per-batch: M = Tk, N = 2C, K = C); epilogue
  // scatters K transposed and V head-major.
  gemm_bf16_wmma<EPI_KV><<<dim3((2 * C) / 128, Tk / 128, B), 256, 0, stream>>>(
      cw, wkvb, bkv, nullptr, nullptr, Kt, Vp, 2 * C, C, C, 2 * C,
      (size_t)Tk * C);

  // attention
  flash_attn_wmma<<<dim3(Tq / 64, B * H, 1), 128, 0, stream>>>(Qp, Kt, Vp, attn);

  // out = attn @ Wo + bo  (f32 output)
  gemm_bf16_wmma<EPI_O><<<dim3(C / 128, (B * Tq) / 128, 1), 256, 0, stream>>>(
      attn, wob, bo, nullptr, out, nullptr, nullptr, C, C, C, C, 0);
}
